// EdgeConvBlock_22926535426431
// MI455X (gfx1250) — matrix operand order repaired and checked by reading, not compiled, not run on hardware
//
#include <hip/hip_runtime.h>

typedef _Float16 h16;
typedef __attribute__((ext_vector_type(4)))  _Float16 v4h;
typedef __attribute__((ext_vector_type(8)))  _Float16 v8h;
typedef __attribute__((ext_vector_type(16))) _Float16 v16h;
typedef __attribute__((ext_vector_type(4)))  float    v4f;
typedef __attribute__((ext_vector_type(8)))  float    v8f;

#define WPB       8            // waves per block
#define THREADS   (WPB * 32)
#define W1_STRIDE 136          // 128 + 8 halves pad -> 4-bank shift per row
#define W2_STRIDE 72           // 64 + 8 halves pad
#define F_STRIDE  136
#define H_STRIDE  72
#define W2T_HALVES  (64 * W2_STRIDE)                     // 4608, persistent
// pool reused: phase1 W1^T staging (64*136=8704) vs phase3 feat+h buffers
#define POOL_HALVES (WPB * 16 * (F_STRIDE + H_STRIDE))   // 26624

__device__ __forceinline__ v16h lds_frag16(const h16* p) {
  v8h lo = *(const v8h*)(p);
  v8h hh = *(const v8h*)(p + 8);
  return __builtin_shufflevector(lo, hh, 0, 1, 2, 3, 4, 5, 6, 7,
                                         8, 9, 10, 11, 12, 13, 14, 15);
}

__device__ __forceinline__ void wave_sync_lds() {
  __builtin_amdgcn_wave_barrier();
  asm volatile("s_wait_dscnt 0x0" ::: "memory");
  __builtin_amdgcn_wave_barrier();
}

// WGP-scope prefetch (SCOPE 0 = default): pulls line into all cache levels,
// so next iteration's staging loads hit the WGP-local cache, not just GL2.
// Addresses are always valid (clamped node rows) -> non-speculative is safe.
__device__ __forceinline__ void prefetch_wgp(const void* p) {
  asm volatile("global_prefetch_b8 %0, off" :: "v"((unsigned long long)p));
}

// monotonic float <-> uint order map, 2 VALU ops: u ^ ((i>>31)|0x80000000)
__device__ __forceinline__ unsigned int f32_order(float v) {
  unsigned int u = __float_as_uint(v);
  return u ^ (unsigned int)(((int)u >> 31) | (int)0x80000000);
}

__device__ __forceinline__ long edge_clamp(long tt, int er, int E) {
  long e = tt * 16 + er;
  return (e >= (long)E) ? ((long)E - 1) : e;
}

__global__ void edgeconv_init_kernel(unsigned int* __restrict__ out, int n) {
  int i = blockIdx.x * blockDim.x + threadIdx.x;
  if (i < n) out[i] = 0x007FFFFFu;  // f32_order(-inf)
}

__global__ void edgeconv_finalize_kernel(float* __restrict__ out, int n) {
  int i = blockIdx.x * blockDim.x + threadIdx.x;
  if (i < n) {
    unsigned int u = ((const unsigned int*)out)[i];
    unsigned int fb = u ^ (unsigned int)(((int)~u >> 31) | (int)0x80000000);
    float v = __uint_as_float(fb);
    bool finite = (v >= -3.4e38f) && (v <= 3.4e38f);  // false for +/-inf, NaN
    out[i] = finite ? v : 0.0f;
  }
}

__global__ __launch_bounds__(THREADS)
void edgeconv_tile_kernel(const float* __restrict__ x,
                          const long long* __restrict__ ei,   // [2, E] int64
                          const float* __restrict__ W1,       // [128, 64]
                          const float* __restrict__ b1,       // [64]
                          const float* __restrict__ W2,       // [64, 64]
                          const float* __restrict__ b2,       // [64]
                          unsigned int* __restrict__ out,     // [N, 64] order-mapped
                          int E) {
  __shared__ __align__(16) h16 sW2T[W2T_HALVES];  // persistent W2^T (f16)
  __shared__ __align__(16) h16 smem[POOL_HALVES]; // W1 staging / feat+h pool
  __shared__ int sDst[WPB][16];

  const int tid  = threadIdx.x;
  const int wave = tid >> 5;
  const int lane = tid & 31;
  const int r16  = lane & 15;   // N-column / M-row selector per WMMA layout
  const int hi   = lane >> 4;   // lane half
  const int er   = lane >> 1;   // staging: edge row 0..15 (2 lanes per row)
  const int hs   = lane & 1;    // staging: 32-column chunk

  // ================= phase 1: stage weights (transposed, f16) ==============
  h16* sW1T = smem;  // 64 x W1_STRIDE (pool reused later)
  for (int i = tid; i < 128 * 64; i += THREADS) {
    int k = i >> 6, n = i & 63;
    sW1T[n * W1_STRIDE + k] = (h16)W1[i];
  }
  for (int i = tid; i < 64 * 64; i += THREADS) {
    int k = i >> 6, n = i & 63;
    sW2T[n * W2_STRIDE + k] = (h16)W2[i];
  }
  __syncthreads();

  // ====== phase 2: hoist W1 B-fragments into registers (loop-invariant) ====
  v16h w1F[4][4];   // [nt][kt]  128 VGPRs
#pragma unroll
  for (int nt = 0; nt < 4; ++nt)
#pragma unroll
    for (int kt = 0; kt < 4; ++kt)
      w1F[nt][kt] = lds_frag16(&sW1T[(nt * 16 + r16) * W1_STRIDE + kt * 32 + hi * 16]);
  float b1v[4], b2v[4];
#pragma unroll
  for (int nt = 0; nt < 4; ++nt) {
    b1v[nt] = b1[nt * 16 + r16];
    b2v[nt] = b2[nt * 16 + r16];
  }
  asm volatile("s_wait_dscnt 0x0" ::: "memory");  // fragments in VGPRs before reuse
  __syncthreads();

  // ================= phase 3: prefetch-pipelined tile loop =================
  h16* sFeat = smem + wave * (16 * F_STRIDE);
  h16* sHbuf = smem + WPB * 16 * F_STRIDE + wave * (16 * H_STRIDE);

  const int stride   = gridDim.x * WPB;
  const int numTiles = (E + 15) >> 4;
  int t = blockIdx.x * WPB + wave;
  if (t >= numTiles) return;

  // two-deep edge-index pipeline: (sA,dA) tile t, (sB,dB) tile t+stride
  long e0 = edge_clamp(t, er, E);
  long sA = (long)ei[e0], dA = (long)ei[(long)E + e0];
  long e1 = edge_clamp((long)t + stride, er, E);
  long sB = (long)ei[e1], dB = (long)ei[(long)E + e1];

  for (; t < numTiles; t += stride) {
    // -- 1. stage tile t: [x_i | x_j - x_i] as f16 (rows prefetched last iter) --
    {
      if (hs == 0) sDst[wave][er] = (int)dA;
      const float* xd = x + dA * 64 + hs * 32;
      const float* xs = x + sA * 64 + hs * 32;
      h16* frow = sFeat + er * F_STRIDE + hs * 32;
#pragma unroll
      for (int c = 0; c < 32; c += 4) {
        v4f xi = *(const v4f*)(xd + c);
        v4f xj = *(const v4f*)(xs + c);
        *(v4h*)(frow + c)      = __builtin_convertvector(xi, v4h);       // x_i
        *(v4h*)(frow + 64 + c) = __builtin_convertvector(xj - xi, v4h);  // x_j-x_i
      }
    }
    // -- 2. WGP-scope prefetch of next tile's x rows (128B per pointer);
    //       load edge indices for t+2*stride (two-deep pipeline) --
    {
      const float* pxd = x + dB * 64 + hs * 32;
      const float* pxs = x + sB * 64 + hs * 32;
      prefetch_wgp(pxd);
      prefetch_wgp(pxd + 16);
      prefetch_wgp(pxs);
      prefetch_wgp(pxs + 16);
    }
    long e2 = edge_clamp((long)t + 2 * (long)stride, er, E);
    long sC = (long)ei[e2], dC = (long)ei[(long)E + e2];
    wave_sync_lds();

    // -- 3. GEMM1: [16x128] @ W1; A from LDS, B from registers --
    v16h aF[4];
    {
      const h16* abase = sFeat + r16 * F_STRIDE;
#pragma unroll
      for (int kt = 0; kt < 4; ++kt) {
        v8h lo = *(const v8h*)(abase + kt * 32 + hi * 16);
        v8h hh = *(const v8h*)(abase + kt * 32 + hi * 16 + 8);
        aF[kt] = __builtin_shufflevector(lo, hh, 0, 1, 2, 3, 4, 5, 6, 7,
                                                 8, 9, 10, 11, 12, 13, 14, 15);
      }
    }
    v8f acc[4] = {};
#pragma unroll
    for (int kt = 0; kt < 4; ++kt) {
#pragma unroll
      for (int nt = 0; nt < 4; ++nt) {
        acc[nt] = __builtin_amdgcn_wmma_f32_16x16x32_f16(
            false, aF[kt], false, w1F[nt][kt], (short)0, acc[nt], false, false);
      }
    }

    // -- 4. bias + relu -> sHbuf; GEMM2: [16x64] @ W2 (B from LDS) --
#pragma unroll
    for (int nt = 0; nt < 4; ++nt) {
#pragma unroll
      for (int v = 0; v < 8; ++v) {
        float hv = acc[nt][v] + b1v[nt];
        hv = hv > 0.0f ? hv : 0.0f;
        sHbuf[(v + 8 * hi) * H_STRIDE + nt * 16 + r16] = (h16)hv;
      }
    }
    wave_sync_lds();
    v16h a2[2];
    {
      const h16* abase = sHbuf + r16 * H_STRIDE;
#pragma unroll
      for (int kt = 0; kt < 2; ++kt) {
        v8h lo = *(const v8h*)(abase + kt * 32 + hi * 16);
        v8h hh = *(const v8h*)(abase + kt * 32 + hi * 16 + 8);
        a2[kt] = __builtin_shufflevector(lo, hh, 0, 1, 2, 3, 4, 5, 6, 7,
                                                 8, 9, 10, 11, 12, 13, 14, 15);
      }
    }
    v8f acc2[4] = {};
#pragma unroll
    for (int kt = 0; kt < 2; ++kt) {
#pragma unroll
      for (int nt = 0; nt < 4; ++nt) {
        v16h bF = lds_frag16(&sW2T[(nt * 16 + r16) * W2_STRIDE + kt * 32 + hi * 16]);
        acc2[nt] = __builtin_amdgcn_wmma_f32_16x16x32_f16(
            false, a2[kt], false, bF, (short)0, acc2[nt], false, false);
      }
    }

    // -- 5. scatter-max: one branch-free u32 atomic per output element --
#pragma unroll
    for (int nt = 0; nt < 4; ++nt) {
      const int n = nt * 16 + r16;
#pragma unroll
      for (int v = 0; v < 8; ++v) {
        const int node = sDst[wave][v + 8 * hi];
        atomicMax(out + (long)node * 64 + n, f32_order(acc2[nt][v] + b2v[nt]));
      }
    }

    sA = sB; dA = dB; sB = sC; dB = dC;   // advance index pipeline
    wave_sync_lds();  // all sFeat/sDst reads done before next iter overwrites
  }
}

extern "C" void kernel_launch(void* const* d_in, const int* in_sizes, int n_in,
                              void* d_out, int out_size, void* d_ws, size_t ws_size,
                              hipStream_t stream) {
  const float*     x  = (const float*)d_in[0];
  const long long* ei = (const long long*)d_in[1];  // int64 [2, E]
  const float*     W1 = (const float*)d_in[2];
  const float*     b1 = (const float*)d_in[3];
  const float*     W2 = (const float*)d_in[4];
  const float*     b2 = (const float*)d_in[5];
  float* out = (float*)d_out;

  const int E = in_sizes[1] / 2;
  const int outN = out_size;  // N_NODES * 64

  const int gInit = (outN + 255) / 256;
  edgeconv_init_kernel<<<gInit, 256, 0, stream>>>((unsigned int*)out, outN);

  const int numTiles = (E + 15) >> 4;
  int blocks = (numTiles + WPB - 1) / WPB;
  if (blocks > 2048) blocks = 2048;
  edgeconv_tile_kernel<<<blocks, THREADS, 0, stream>>>(x, ei, W1, b1, W2, b2,
                                                       (unsigned int*)out, E);

  edgeconv_finalize_kernel<<<gInit, 256, 0, stream>>>(out, outN);
}
